// SwinTransformerV2Block_24077586661777
// MI455X (gfx1250) — compile-verified, matrix operand order checked by reading, not run on hardware
//
#include <hip/hip_runtime.h>

typedef __attribute__((ext_vector_type(16))) _Float16 v16h;
typedef __attribute__((ext_vector_type(8)))  float    v8f;
typedef __attribute__((ext_vector_type(4)))  unsigned int u32x4;
typedef __attribute__((ext_vector_type(8)))  int i32x8;
typedef __attribute__((ext_vector_type(4)))  int i32x4;

#if defined(__has_builtin)
#if __has_builtin(__builtin_amdgcn_tensor_load_to_lds) && __has_builtin(__builtin_amdgcn_s_wait_tensorcnt)
#define HAVE_TDM 1
#endif
#endif

constexpr int kSS   = 4;
constexpr int kDim  = 384;
constexpr int kNH   = 12;
constexpr int kHD   = 32;
constexpr int kNTok = 512;
constexpr int kNWin = 16;                 // B * 8 windows
constexpr int kHid  = 1536;
constexpr int kMTot = kNWin * kNTok;      // 8192 window-tokens
constexpr int kTblRows = 15 * 15 * 15;    // 3375
constexpr float kLnEps = 1e-6f;

// dynamic-LDS layout for the attention kernel
constexpr int kLdsKBytes  = kNTok * kHD * 2;              // 32768: K panel (TDM dest)
constexpr int kVtStride   = kNTok + 8;                    // halves, keeps 16B alignment
constexpr int kLdsVtBytes = kHD * kVtStride * 2;          // 33280: transposed V
constexpr int kLdsPbBytes = 8 * 16 * 32 * 2;              // 8192: per-wave P staging
constexpr int kLdsTotal   = kLdsKBytes + kLdsVtBytes + kLdsPbBytes;  // 74240

union FragH { v16h v; float4 f4[2]; };

__device__ __forceinline__ v8f wmma16(v16h a, v16h b, v8f c) {
  return __builtin_amdgcn_wmma_f32_16x16x32_f16(false, a, false, b, (short)0, c, false, false);
}
__device__ __forceinline__ int region(int g) { return g < 8 ? 0 : (g < 12 ? 1 : 2); }
__device__ __forceinline__ float redmax16(float x) {
#pragma unroll
  for (int m = 1; m < 16; m <<= 1) x = fmaxf(x, __shfl_xor(x, m, 16));
  return x;
}
__device__ __forceinline__ float redsum16(float x) {
#pragma unroll
  for (int m = 1; m < 16; m <<= 1) x += __shfl_xor(x, m, 16);
  return x;
}

// ---------------- small prep kernels ----------------

__global__ void k_cast_f32_f16(const float* __restrict__ s, _Float16* __restrict__ d, int n) {
  int i = blockIdx.x * blockDim.x + threadIdx.x;
  if (i < n) d[i] = (_Float16)s[i];
}

__global__ void k_qkv_bias(const float* __restrict__ qb, const float* __restrict__ vb,
                           float* __restrict__ o) {
  int i = blockIdx.x * blockDim.x + threadIdx.x;
  if (i >= 3 * kDim) return;
  o[i] = (i < kDim) ? qb[i] : ((i < 2 * kDim) ? 0.f : vb[i - 2 * kDim]);
}

// cyclic shift + window partition + f16 cast: x (2,4096,384) -> xw (16,512,384)
__global__ void k_shift(const float* __restrict__ x, _Float16* __restrict__ xw) {
  int i = blockIdx.x * blockDim.x + threadIdx.x;
  if (i >= kMTot * kDim) return;
  int c = i % kDim, m = i / kDim;
  int w = m >> 9, tok = m & 511;
  int b_ = w >> 3, widx = w & 7;
  int wh = (widx >> 2) & 1, wy = (widx >> 1) & 1, wt = widx & 1;
  int hh = (tok >> 6) & 7, lw = (tok >> 3) & 7, lt = tok & 7;
  int gh = (wh * 8 + hh + kSS) & 15;
  int gw = (wy * 8 + lw + kSS) & 15;
  int gt = (wt * 8 + lt + kSS) & 15;
  xw[i] = (_Float16)x[((size_t)(b_ * 4096 + gh * 256 + gw * 16 + gt)) * kDim + c];
}

// continuous position bias MLP -> btab[3375][12] = 16*sigmoid(rpb)
__global__ void k_cpb(const float* __restrict__ w1, const float* __restrict__ b1,
                      const float* __restrict__ w2, float* __restrict__ btab) {
  int r = blockIdx.x * blockDim.x + threadIdx.x;
  if (r >= kTblRows) return;
  int ids[3] = { r / 225, (r / 15) % 15, r % 15 };
  float cc[3];
#pragma unroll
  for (int a = 0; a < 3; ++a) {
    float t = (float)(ids[a] - 7) * (8.0f / 7.0f);
    float s = (t > 0.f) ? 1.f : ((t < 0.f) ? -1.f : 0.f);
    cc[a] = s * log2f(fabsf(t) + 1.f) * (1.f / 3.f);   // /log2(8)
  }
  float acc[kNH];
#pragma unroll
  for (int hd = 0; hd < kNH; ++hd) acc[hd] = 0.f;
  for (int hh = 0; hh < 512; ++hh) {
    float hv = cc[0] * w1[hh * 3 + 0] + cc[1] * w1[hh * 3 + 1] + cc[2] * w1[hh * 3 + 2] + b1[hh];
    hv = fmaxf(hv, 0.f);
#pragma unroll
    for (int hd = 0; hd < kNH; ++hd) acc[hd] += hv * w2[hd * 512 + hh];
  }
#pragma unroll
  for (int hd = 0; hd < kNH; ++hd)
    btab[r * kNH + hd] = 16.f / (1.f + __expf(-acc[hd]));
}

// ------- generic WMMA GEMM: C = A(MxK,f16) @ W(NxK,f16)^T + bias, 64x16 per wave -------
// mode bit0: f16 output, bit1: exact GELU, bit2: window-reverse scatter (f32 output)
__global__ __launch_bounds__(256) void k_gemm(
    const _Float16* __restrict__ A, const _Float16* __restrict__ W,
    const float* __restrict__ bias, float* __restrict__ outf,
    _Float16* __restrict__ outh, int M, int N, int K, int mode) {
  const int gwv = (int)((blockIdx.x * blockDim.x + threadIdx.x) >> 5);
  const int ntile = N >> 4;
  const int tiles = (M >> 6) * ntile;
  if (gwv >= tiles) return;
  const int m0 = (gwv / ntile) << 6;     // 64-row M block
  const int n0 = (gwv % ntile) << 4;
  const int lane = threadIdx.x & 31, ln = lane & 15, hf = lane >> 4;
  const _Float16* a0 = A + (size_t)(m0 + ln) * K + hf * 8;
  const _Float16* brow = W + (size_t)(n0 + ln) * K + hf * 8;
  const size_t rstep = (size_t)16 * K;
  v8f acc[4] = { {}, {}, {}, {} };
  for (int kk = 0; kk < K; kk += 32) {
    FragH b;
    b.f4[0] = *(const float4*)(brow + kk);
    b.f4[1] = *(const float4*)(brow + kk + 16);
    __builtin_prefetch(brow + kk + 128, 0, 1);   // global_prefetch_b8 on the weight stream
#pragma unroll
    for (int s = 0; s < 4; ++s) {
      FragH a;
      a.f4[0] = *(const float4*)(a0 + (size_t)s * rstep + kk);
      a.f4[1] = *(const float4*)(a0 + (size_t)s * rstep + kk + 16);
      acc[s] = wmma16(a.v, b.v, acc[s]);
    }
  }
  const int n = n0 + ln;
  const float bn = bias[n];
#pragma unroll
  for (int s = 0; s < 4; ++s) {
#pragma unroll
    for (int v = 0; v < 8; ++v) {
      const int m = m0 + s * 16 + v + hf * 8;
      float val = acc[s][v] + bn;
      if (mode & 2) val = 0.5f * val * (1.f + erff(val * 0.70710678f));
      if (mode & 4) {
        const int wwin = m >> 9, tok = m & 511;
        const int b_ = wwin >> 3, widx = wwin & 7;
        const int wh = (widx >> 2) & 1, wy = (widx >> 1) & 1, wt = widx & 1;
        const int hh = (tok >> 6) & 7, lw = (tok >> 3) & 7, lt = tok & 7;
        const int gh = (wh * 8 + hh + kSS) & 15;
        const int gw2 = (wy * 8 + lw + kSS) & 15;
        const int gt = (wt * 8 + lt + kSS) & 15;
        outf[((size_t)(b_ * 4096 + gh * 256 + gw2 * 16 + gt)) * kDim + n] = val;
      } else if (mode & 1) {
        outh[(size_t)m * N + n] = (_Float16)val;
      } else {
        outf[(size_t)m * N + n] = val;
      }
    }
  }
}

// cosine-normalize q,k per (window,head,token); split qkv -> qn/kn/v f16 head-major
__global__ void k_normcast(const _Float16* __restrict__ qkv, _Float16* __restrict__ qn,
                           _Float16* __restrict__ kn, _Float16* __restrict__ vv) {
  int t = blockIdx.x * blockDim.x + threadIdx.x;
  if (t >= kNWin * kNH * kNTok) return;
  int w = t / (kNH * kNTok);
  int rem = t % (kNH * kNTok);
  int h = rem / kNTok, tok = rem % kNTok;
  const _Float16* base = qkv + ((size_t)(w * kNTok + tok)) * (3 * kDim);
  const _Float16* qp = base + h * kHD;
  const _Float16* kp = base + kDim + h * kHD;
  const _Float16* vp = base + 2 * kDim + h * kHD;
  size_t o = (((size_t)(w * kNH + h)) * kNTok + tok) * kHD;
  float sq = 0.f, sk = 0.f;
  for (int d = 0; d < kHD; ++d) {
    float q = (float)qp[d]; sq += q * q;
    float k = (float)kp[d]; sk += k * k;
  }
  float iq = 1.f / fmaxf(sqrtf(sq), 1e-12f);
  float ik = 1.f / fmaxf(sqrtf(sk), 1e-12f);
  for (int d = 0; d < kHD; ++d) {
    qn[o + d] = (_Float16)((float)qp[d] * iq);
    kn[o + d] = (_Float16)((float)kp[d] * ik);
    vv[o + d] = vp[d];
  }
}

// ---------------- flash-style windowed attention, one (window,head) per workgroup ----------------
__global__ __launch_bounds__(256) void k_attn(
    const _Float16* __restrict__ qn, const _Float16* __restrict__ kn,
    const _Float16* __restrict__ vv, const float* __restrict__ btab,
    const float* __restrict__ logit_scale, _Float16* __restrict__ ao) {
  extern __shared__ char smem[];
  _Float16* kst = (_Float16*)smem;                                  // 512x32 K panel
  _Float16* vT  = (_Float16*)(smem + kLdsKBytes);                   // 32 x (512+8) transposed V
  _Float16* Pb  = (_Float16*)(smem + kLdsKBytes + kLdsVtBytes);     // 8 waves x 16x32 P staging

  const int w = blockIdx.x / kNH;
  const int h = blockIdx.x % kNH;
  const int widx = w & 7;
  const int tid = threadIdx.x;

  const size_t head_off = ((size_t)(w * kNH + h)) * kNTok * kHD;
  const _Float16* qb = qn + head_off;
  const _Float16* kb = kn + head_off;
  const _Float16* vb = vv + head_off;

#if HAVE_TDM
  if (tid < 32) {
    // Tensor Data Mover: 1-D tile, 4096 elements x 8B = 32KB, global kb -> LDS kst.
    // D# group0: count=1 (user), lds_addr, global_addr[56:0], type=2.
    const unsigned lds_off = (unsigned)(size_t)(void*)kst;
    const unsigned long long ga = (unsigned long long)(size_t)(const void*)kb;
    u32x4 g0 = {};
    g0.x = 1u;                                          // count=1, is_restore=0
    g0.y = lds_off;                                     // lds_addr
    g0.z = (unsigned)(ga & 0xffffffffu);                // global_addr[31:0]
    g0.w = (unsigned)((ga >> 32) & 0x01ffffffu) | (2u << 30);  // global_addr[56:32], type=2
    // D# group1: data_size=3 (8B); tensor_dim0=4096; tensor_dim1=1; tile_dim0=4096;
    // tile_dim1/2 unused; tensor_dim0_stride=4096.
    i32x8 g1 = {};
    g1[0] = (int)(3u << 16);                            // workgroup_mask=0, data_size=8B
    g1[1] = (int)(4096u << 16);                         // tensor_dim0[15:0] @ bits 63:48
    g1[2] = (int)(1u << 16);                            // tensor_dim0 hi=0; tensor_dim1 lo=1
    g1[3] = (int)(4096u << 16);                         // tensor_dim1 hi=0; tile_dim0=4096
    g1[4] = 0;                                          // tile_dim1=0, tile_dim2=0
    g1[5] = 4096;                                       // tensor_dim0_stride lo
    g1[6] = 0;
    g1[7] = 0;
    i32x4 z4 = {};
    i32x8 z8 = {};
    __builtin_amdgcn_tensor_load_to_lds(g0, g1, z4, z4, z8, 0);
    __builtin_amdgcn_s_wait_tensorcnt(0);
  }
#else
  for (int i = tid; i < kNTok * kHD; i += 256) kst[i] = kb[i];
#endif

  for (int i = tid; i < kNTok * kHD; i += 256) {
    int key = i >> 5, d = i & 31;
    vT[d * kVtStride + key] = vb[i];
  }
  __syncthreads();

  const float scale = __expf(fminf(logit_scale[h], 4.6051702f));  // clamp at ln(100)
  const int wave = tid >> 5, lane = tid & 31, ln = lane & 15, hf = lane >> 4;
  const int wh = (widx >> 2) & 1, wy = (widx >> 1) & 1, wt = widx & 1;

  for (int qt = wave; qt < kNTok / 16; qt += 8) {
    FragH qa;
    {
      const _Float16* qr = qb + (size_t)(qt * 16 + ln) * kHD + hf * 8;
      qa.f4[0] = *(const float4*)qr;
      qa.f4[1] = *(const float4*)(qr + 16);
    }
    int qh[8], qwv[8], qlv[8], qlab[8];
#pragma unroll
    for (int v = 0; v < 8; ++v) {
      int m = qt * 16 + v + hf * 8;
      qh[v] = (m >> 6) & 7; qwv[v] = (m >> 3) & 7; qlv[v] = m & 7;
      qlab[v] = region(wh * 8 + qh[v]) * 9 + region(wy * 8 + qwv[v]) * 3 +
                region(wt * 8 + qlv[v]);
    }
    float Mr[8], Lr[8];
#pragma unroll
    for (int v = 0; v < 8; ++v) { Mr[v] = -1e30f; Lr[v] = 0.f; }
    v8f O0 = {}, O1 = {};

    for (int cj = 0; cj < 16; ++cj) {
      const int k0 = cj * 32;
      FragH bk0, bk1;
      {
        const _Float16* kr = kst + (size_t)(k0 + ln) * kHD + hf * 8;
        bk0.f4[0] = *(const float4*)kr; bk0.f4[1] = *(const float4*)(kr + 16);
      }
      {
        const _Float16* kr = kst + (size_t)(k0 + 16 + ln) * kHD + hf * 8;
        bk1.f4[0] = *(const float4*)kr; bk1.f4[1] = *(const float4*)(kr + 16);
      }
      v8f z = {};
      v8f s1 = wmma16(qa.v, bk0.v, z);
      v8f s2 = wmma16(qa.v, bk1.v, z);

      const int kA = k0 + ln, kB = k0 + 16 + ln;
      const int kah = (kA >> 6) & 7, kaw = (kA >> 3) & 7, kal = kA & 7;
      const int kbh = (kB >> 6) & 7, kbw = (kB >> 3) & 7, kbl = kB & 7;
      const int klA = region(wh * 8 + kah) * 9 + region(wy * 8 + kaw) * 3 + region(wt * 8 + kal);
      const int klB = region(wh * 8 + kbh) * 9 + region(wy * 8 + kbw) * 3 + region(wt * 8 + kbl);

      float p1[8], p2[8];
#pragma unroll
      for (int v = 0; v < 8; ++v) {
        int ia = (((qh[v] - kah + 7) * 15 + (qwv[v] - kaw + 7)) * 15 + (qlv[v] - kal + 7)) * kNH + h;
        int ib = (((qh[v] - kbh + 7) * 15 + (qwv[v] - kbw + 7)) * 15 + (qlv[v] - kbl + 7)) * kNH + h;
        p1[v] = s1[v] * scale + btab[ia] + (qlab[v] == klA ? 0.f : -100.f);
        p2[v] = s2[v] * scale + btab[ib] + (qlab[v] == klB ? 0.f : -100.f);
      }
#pragma unroll
      for (int v = 0; v < 8; ++v) {
        float mx = redmax16(fmaxf(p1[v], p2[v]));
        float nm = fmaxf(Mr[v], mx);
        float alpha = __expf(Mr[v] - nm);
        p1[v] = __expf(p1[v] - nm);
        p2[v] = __expf(p2[v] - nm);
        float rs = redsum16(p1[v] + p2[v]);
        Lr[v] = Lr[v] * alpha + rs;
        Mr[v] = nm;
        O0[v] *= alpha;
        O1[v] *= alpha;
        Pb[wave * 512 + (v + hf * 8) * 32 + ln] = (_Float16)p1[v];
        Pb[wave * 512 + (v + hf * 8) * 32 + 16 + ln] = (_Float16)p2[v];
      }
      asm volatile("s_wait_dscnt 0" ::: "memory");   // wave-local LDS RAW
      FragH pa, bv0, bv1;
      {
        const _Float16* pr = Pb + wave * 512 + ln * 32 + hf * 8;
        pa.f4[0] = *(const float4*)pr; pa.f4[1] = *(const float4*)(pr + 16);
      }
      {
        const _Float16* vr = vT + ln * kVtStride + k0 + hf * 8;
        bv0.f4[0] = *(const float4*)vr; bv0.f4[1] = *(const float4*)(vr + 16);
      }
      {
        const _Float16* vr = vT + (16 + ln) * kVtStride + k0 + hf * 8;
        bv1.f4[0] = *(const float4*)vr; bv1.f4[1] = *(const float4*)(vr + 16);
      }
      O0 = wmma16(pa.v, bv0.v, O0);
      O1 = wmma16(pa.v, bv1.v, O1);
    }
#pragma unroll
    for (int v = 0; v < 8; ++v) {
      float inv = 1.f / Lr[v];
      int m = qt * 16 + v + hf * 8;
      size_t o = ((size_t)w * kNTok + m) * kDim + h * kHD;
      ao[o + ln]      = (_Float16)(O0[v] * inv);
      ao[o + 16 + ln] = (_Float16)(O1[v] * inv);
    }
  }
}

// x1 = shortcut + LN(y) ; also emit f16 copy for the MLP GEMM
__global__ void k_resln1(const float* __restrict__ x, const float* __restrict__ y,
                         const float* __restrict__ g, const float* __restrict__ b,
                         float* __restrict__ x1, _Float16* __restrict__ x1h) {
  int t = blockIdx.x * blockDim.x + threadIdx.x;
  if (t >= kMTot) return;
  const float* yr = y + (size_t)t * kDim;
  const float* xr = x + (size_t)t * kDim;
  float mu = 0.f;
  for (int c = 0; c < kDim; ++c) mu += yr[c];
  mu *= (1.f / kDim);
  float var = 0.f;
  for (int c = 0; c < kDim; ++c) { float d = yr[c] - mu; var += d * d; }
  var *= (1.f / kDim);
  float inv = rsqrtf(var + kLnEps);
  for (int c = 0; c < kDim; ++c) {
    float v = xr[c] + ((yr[c] - mu) * inv * g[c] + b[c]);
    x1[(size_t)t * kDim + c] = v;
    x1h[(size_t)t * kDim + c] = (_Float16)v;
  }
}

// out = x1 + LN(tf) with gamma=1, beta=0
__global__ void k_resln2(const float* __restrict__ x1, const float* __restrict__ tf,
                         float* __restrict__ out) {
  int t = blockIdx.x * blockDim.x + threadIdx.x;
  if (t >= kMTot) return;
  const float* yr = tf + (size_t)t * kDim;
  float mu = 0.f;
  for (int c = 0; c < kDim; ++c) mu += yr[c];
  mu *= (1.f / kDim);
  float var = 0.f;
  for (int c = 0; c < kDim; ++c) { float d = yr[c] - mu; var += d * d; }
  var *= (1.f / kDim);
  float inv = rsqrtf(var + kLnEps);
  for (int c = 0; c < kDim; ++c)
    out[(size_t)t * kDim + c] = x1[(size_t)t * kDim + c] + (yr[c] - mu) * inv;
}

// ---------------- host launcher ----------------

extern "C" void kernel_launch(void* const* d_in, const int* in_sizes, int n_in,
                              void* d_out, int out_size, void* d_ws, size_t ws_size,
                              hipStream_t stream) {
  (void)in_sizes; (void)n_in; (void)out_size; (void)ws_size;
  const float* x        = (const float*)d_in[0];
  const float* qkv_w    = (const float*)d_in[1];
  const float* q_bias   = (const float*)d_in[2];
  const float* v_bias   = (const float*)d_in[3];
  const float* logit_sc = (const float*)d_in[4];
  const float* cpb_w1   = (const float*)d_in[5];
  const float* cpb_b1   = (const float*)d_in[6];
  const float* cpb_w2   = (const float*)d_in[7];
  const float* proj_w   = (const float*)d_in[8];
  const float* proj_b   = (const float*)d_in[9];
  const float* norm1_g  = (const float*)d_in[10];
  const float* norm1_b  = (const float*)d_in[11];
  const float* mlp_w1   = (const float*)d_in[12];
  const float* mlp_b1   = (const float*)d_in[13];
  const float* mlp_w2   = (const float*)d_in[14];
  const float* mlp_b2   = (const float*)d_in[15];
  float* out = (float*)d_out;

  char* ws = (char*)d_ws;
  size_t off = 0;
  auto take = [&](size_t bytes) -> void* {
    void* p = ws + off;
    off = (off + bytes + 255) & ~(size_t)255;
    return p;
  };
  _Float16* qkvw_h  = (_Float16*)take((size_t)3 * kDim * kDim * 2);
  _Float16* projw_h = (_Float16*)take((size_t)kDim * kDim * 2);
  _Float16* w1_h    = (_Float16*)take((size_t)kHid * kDim * 2);
  _Float16* w2_h    = (_Float16*)take((size_t)kDim * kHid * 2);
  float*    qkvb    = (float*)take((size_t)3 * kDim * 4);
  _Float16* xw_h    = (_Float16*)take((size_t)kMTot * kDim * 2);
  float*    btab    = (float*)take((size_t)kTblRows * kNH * 4);
  _Float16* qkv_h   = (_Float16*)take((size_t)kMTot * 3 * kDim * 2);
  _Float16* qn_h    = (_Float16*)take((size_t)kNWin * kNH * kNTok * kHD * 2);
  _Float16* kn_h    = (_Float16*)take((size_t)kNWin * kNH * kNTok * kHD * 2);
  _Float16* v_h     = (_Float16*)take((size_t)kNWin * kNH * kNTok * kHD * 2);
  _Float16* ao_h    = (_Float16*)take((size_t)kMTot * kDim * 2);
  float*    y_f     = (float*)take((size_t)kMTot * kDim * 4);
  float*    x1_f    = (float*)take((size_t)kMTot * kDim * 4);
  _Float16* x1_h    = (_Float16*)take((size_t)kMTot * kDim * 2);
  float*    t_f     = (float*)take((size_t)kMTot * kDim * 4);
  _Float16* hdn_h   = (_Float16*)take((size_t)kMTot * kHid * 2);

  const int T = 256;
  auto blk = [](long long n, int t) { return (int)((n + t - 1) / t); };

  // weight casts + fused qkv bias
  k_cast_f32_f16<<<blk(3LL * kDim * kDim, T), T, 0, stream>>>(qkv_w, qkvw_h, 3 * kDim * kDim);
  k_cast_f32_f16<<<blk((long long)kDim * kDim, T), T, 0, stream>>>(proj_w, projw_h, kDim * kDim);
  k_cast_f32_f16<<<blk((long long)kHid * kDim, T), T, 0, stream>>>(mlp_w1, w1_h, kHid * kDim);
  k_cast_f32_f16<<<blk((long long)kDim * kHid, T), T, 0, stream>>>(mlp_w2, w2_h, kDim * kHid);
  k_qkv_bias<<<blk(3 * kDim, T), T, 0, stream>>>(q_bias, v_bias, qkvb);

  // shift + window partition ; CPB table
  k_shift<<<blk((long long)kMTot * kDim, T), T, 0, stream>>>(x, xw_h);
  k_cpb<<<blk(kTblRows, T), T, 0, stream>>>(cpb_w1, cpb_b1, cpb_w2, btab);

  // QKV GEMM: M=8192, N=1152, K=384 -> f16
  {
    long long tiles = (kMTot / 64) * ((3 * kDim) / 16);
    k_gemm<<<blk(tiles * 32, T), T, 0, stream>>>(xw_h, qkvw_h, qkvb, nullptr, qkv_h,
                                                 kMTot, 3 * kDim, kDim, 1);
  }
  k_normcast<<<blk((long long)kNWin * kNH * kNTok, T), T, 0, stream>>>(qkv_h, qn_h, kn_h, v_h);

  // attention: one workgroup per (window, head); dynamic LDS (K panel via TDM + V^T + P)
  k_attn<<<kNWin * kNH, T, kLdsTotal, stream>>>(qn_h, kn_h, v_h, btab, logit_sc, ao_h);

  // proj GEMM with window-reverse scatter: M=8192, N=384, K=384 -> y_f (original layout)
  {
    long long tiles = (kMTot / 64) * (kDim / 16);
    k_gemm<<<blk(tiles * 32, T), T, 0, stream>>>(ao_h, projw_h, proj_b, y_f, nullptr,
                                                 kMTot, kDim, kDim, 4);
  }
  k_resln1<<<blk(kMTot, T), T, 0, stream>>>(x, y_f, norm1_g, norm1_b, x1_f, x1_h);

  // MLP1: M=8192, N=1536, K=384, exact GELU -> f16
  {
    long long tiles = (kMTot / 64) * (kHid / 16);
    k_gemm<<<blk(tiles * 32, T), T, 0, stream>>>(x1_h, w1_h, mlp_b1, nullptr, hdn_h,
                                                 kMTot, kHid, kDim, 3);
  }
  // MLP2: M=8192, N=384, K=1536 -> f32
  {
    long long tiles = (kMTot / 64) * (kDim / 16);
    k_gemm<<<blk(tiles * 32, T), T, 0, stream>>>(hdn_h, w2_h, mlp_b2, t_f, nullptr,
                                                 kMTot, kDim, kHid, 0);
  }
  k_resln2<<<blk(kMTot, T), T, 0, stream>>>(x1_f, t_f, out);
}
